// SeqSimCLRLoss_1211180778017
// MI455X (gfx1250) — matrix operand order repaired and checked by reading, not compiled, not run on hardware
//
#include <hip/hip_runtime.h>
#include <math.h>

// ---------------- problem constants ----------------
#define BATCH   512
#define SEQ     25
#define CH      512
#define NWIN    5
#define BS      2560         // BATCH*NWIN
#define NROWS   5120         // 2*BS
#define KDIM    512
#define INV_TEMP 10.0f       // 1/TEMP
#define SHIFT    10.0f       // max possible logit (cos<=1 -> s<=10)
#define NTILE    320         // NROWS/16
#define CT_PER_GROUP 40      // NTILE / 8 waves

#define ALDS_STRIDE 520      // 512 + 8 halfs pad: row stride 260 dwords == 4 mod 64 banks

#define AS1 __attribute__((address_space(1)))
#define AS3 __attribute__((address_space(3)))

typedef __attribute__((ext_vector_type(16))) _Float16 v16h;
typedef __attribute__((ext_vector_type(8)))  _Float16 v8h;
typedef __attribute__((ext_vector_type(8)))  float    v8f;
typedef __attribute__((ext_vector_type(4)))  int      v4i;

// ---------------- kernel 0: zero accumulators ----------------
__global__ void simclr_init_kernel(float* __restrict__ sumexp,
                                   float* __restrict__ pos) {
  int i = blockIdx.x * blockDim.x + threadIdx.x;
  if (i < NROWS) { sumexp[i] = 0.0f; pos[i] = 0.0f; }
}

// ---------------- kernel 1: adaptive pool + L2-normalize + fp16 hi/lo split ----
__global__ __launch_bounds__(256)
void simclr_pool_norm_kernel(const float* __restrict__ zi,
                             const float* __restrict__ zj,
                             _Float16* __restrict__ Zhi,
                             _Float16* __restrict__ Zlo) {
  const int r  = blockIdx.x;
  const float* src = (r < BS) ? zi : zj;
  const int rr = (r < BS) ? r : r - BS;
  const int b  = rr / NWIN;
  const int w  = rr % NWIN;
  const float* base = src + (size_t)(b * SEQ + w * NWIN) * CH;

  const int c0 = threadIdx.x;           // channels c0 and c0+256
  float m0 = 0.f, m1 = 0.f;
#pragma unroll
  for (int t = 0; t < NWIN; ++t) {
    m0 += base[t * CH + c0];
    m1 += base[t * CH + c0 + 256];
  }
  m0 *= 0.2f; m1 *= 0.2f;

  float ss = m0 * m0 + m1 * m1;
#pragma unroll
  for (int m = 16; m >= 1; m >>= 1) ss += __shfl_xor(ss, m, 32);
  __shared__ float red[8];
  const int lane = threadIdx.x & 31, wv = threadIdx.x >> 5;
  if (lane == 0) red[wv] = ss;
  __syncthreads();
  if (threadIdx.x == 0) {
    float tot = 0.f;
#pragma unroll
    for (int i = 0; i < 8; ++i) tot += red[i];
    red[0] = tot;
  }
  __syncthreads();
  const float inv = 1.0f / fmaxf(sqrtf(red[0]), 1e-8f);

  const float v0 = m0 * inv, v1 = m1 * inv;
  const _Float16 h0 = (_Float16)v0;
  const _Float16 h1 = (_Float16)v1;
  const size_t rb = (size_t)r * KDIM;
  Zhi[rb + c0]       = h0;
  Zhi[rb + c0 + 256] = h1;
  Zlo[rb + c0]       = (_Float16)(v0 - (float)h0);
  Zlo[rb + c0 + 256] = (_Float16)(v1 - (float)h1);
}

// ---------------- kernel 2: fused WMMA GEMM + exp-sum + positives ----------
// grid = 320 blocks x 256 threads (8 waves). Block owns row tile rt=blockIdx.x;
// A block (16 rows x 512 K, hi+lo fp16) staged once in LDS via async-to-LDS,
// shared by all 8 waves; wave wv streams col tiles [wv*40, wv*40+40) from L2.
__global__ __launch_bounds__(256)
void simclr_gemm_kernel(const _Float16* __restrict__ Zhi,
                        const _Float16* __restrict__ Zlo,
                        float* __restrict__ sumexp,
                        float* __restrict__ pos) {
  __shared__ _Float16 Ahi_lds[16 * ALDS_STRIDE];
  __shared__ _Float16 Alo_lds[16 * ALDS_STRIDE];

  const int tid  = threadIdx.x;
  const int lane = tid & 31;
  const int wv   = tid >> 5;
  const int rt   = blockIdx.x;          // row tile 0..319
  const int cg   = wv;                  // column group 0..7

  const int laneHalf = lane >> 4;       // 0 or 1
  const int lane15   = lane & 15;

  // ---- stage A block into LDS (contiguous 16KB each for hi/lo) ----
  {
    const _Float16* ghi = Zhi + (size_t)rt * 16 * KDIM;
    const _Float16* glo = Zlo + (size_t)rt * 16 * KDIM;
#pragma unroll
    for (int i = 0; i < 4; ++i) {
      const int c      = tid + i * 256;           // 16-byte chunk id, 0..1023
      const int row    = c >> 6;                  // 64 chunks per 512-half row
      const int col8   = (c & 63) * 8;            // half offset within row
      const int ldsoff = row * ALDS_STRIDE + col8;
      const int goff   = row * KDIM + col8;
#if __has_builtin(__builtin_amdgcn_global_load_async_to_lds_b128)
      __builtin_amdgcn_global_load_async_to_lds_b128(
          (AS1 v4i*)(ghi + goff), (AS3 v4i*)&Ahi_lds[ldsoff], 0, 0);
      __builtin_amdgcn_global_load_async_to_lds_b128(
          (AS1 v4i*)(glo + goff), (AS3 v4i*)&Alo_lds[ldsoff], 0, 0);
#else
      *(v8h*)&Ahi_lds[ldsoff] = *(const v8h*)(ghi + goff);
      *(v8h*)&Alo_lds[ldsoff] = *(const v8h*)(glo + goff);
#endif
    }
#if __has_builtin(__builtin_amdgcn_s_wait_asynccnt)
    __builtin_amdgcn_s_wait_asynccnt(0);
#else
    asm volatile("s_wait_asynccnt 0x0" ::: "memory");
#endif
  }
  __syncthreads();

  // A operand (16x32 f16 layout): row = lane15; lanes>=16 take K+16 block
  const int arow_lds = lane15 * ALDS_STRIDE;
  const int akoff    = laneHalf * 16;

  float rsum[8];
#pragma unroll
  for (int r = 0; r < 8; ++r) rsum[r] = 0.f;

  for (int cti = 0; cti < CT_PER_GROUP; ++cti) {
    const int ct = cg * CT_PER_GROUP + cti;
    // B operand (32x16 f16 layout): col = lane15; lanes>=16 take K+16 block
    const size_t bbase = (size_t)(ct * 16 + lane15) * KDIM;
    const int    bkoff = laneHalf * 16;

    v8f acc = {};
#pragma unroll 4
    for (int k0 = 0; k0 < KDIM; k0 += 32) {
      union { v16h v; v8h h[2]; } ahi, alo;
      ahi.h[0] = *(const v8h*)&Ahi_lds[arow_lds + k0 + akoff];
      ahi.h[1] = *(const v8h*)&Ahi_lds[arow_lds + k0 + akoff + 8];
      alo.h[0] = *(const v8h*)&Alo_lds[arow_lds + k0 + akoff];
      alo.h[1] = *(const v8h*)&Alo_lds[arow_lds + k0 + akoff + 8];
      const v16h bhi = *(const v16h*)(Zhi + bbase + k0 + bkoff);
      const v16h blo = *(const v16h*)(Zlo + bbase + k0 + bkoff);
      // d = hi*hi + hi*lo + lo*hi  (fp32-accurate dot, lo*lo dropped)
      acc = __builtin_amdgcn_wmma_f32_16x16x32_f16(false, ahi.v, false, bhi,
                                                   (short)0, acc, false, false);
      acc = __builtin_amdgcn_wmma_f32_16x16x32_f16(false, ahi.v, false, blo,
                                                   (short)0, acc, false, false);
      acc = __builtin_amdgcn_wmma_f32_16x16x32_f16(false, alo.v, false, bhi,
                                                   (short)0, acc, false, false);
    }

    // epilogue: lane holds col = ct*16+lane15, rows rt*16 + laneHalf*8 + r
    const int gcol     = ct * 16 + lane15;
    const int growbase = rt * 16 + laneHalf * 8;
#pragma unroll
    for (int r = 0; r < 8; ++r) {
      const int   grow = growbase + r;
      const float s    = acc[r] * INV_TEMP;
      if (grow != gcol) rsum[r] += __expf(s - SHIFT);   // diag excluded
      const int partner = (grow < BS) ? grow + BS : grow - BS;
      if (gcol == partner) pos[grow] = s;               // written exactly once
    }
  }

  // reduce each row-sum across its 16-lane half (xor masks stay in-half)
#pragma unroll
  for (int r = 0; r < 8; ++r) {
    float v = rsum[r];
    v += __shfl_xor(v, 1, 32);
    v += __shfl_xor(v, 2, 32);
    v += __shfl_xor(v, 4, 32);
    v += __shfl_xor(v, 8, 32);
    rsum[r] = v;
  }
  if (lane15 == 0) {
    const int rowbase = rt * 16 + laneHalf * 8;
#pragma unroll
    for (int r = 0; r < 8; ++r) atomicAdd(&sumexp[rowbase + r], rsum[r]);
  }
}

// ---------------- kernel 3: scalar loss ----------------
__global__ __launch_bounds__(256)
void simclr_finalize_kernel(const float* __restrict__ sumexp,
                            const float* __restrict__ pos,
                            float* __restrict__ out) {
  float local = 0.f;
  for (int i = threadIdx.x; i < NROWS; i += 256)
    local += SHIFT + logf(sumexp[i]) - pos[i];   // lse_i - pos_i
#pragma unroll
  for (int m = 16; m >= 1; m >>= 1) local += __shfl_xor(local, m, 32);
  __shared__ float red[8];
  const int lane = threadIdx.x & 31, wv = threadIdx.x >> 5;
  if (lane == 0) red[wv] = local;
  __syncthreads();
  if (threadIdx.x == 0) {
    float tot = 0.f;
#pragma unroll
    for (int i = 0; i < 8; ++i) tot += red[i];
    out[0] = tot / (float)NROWS;
  }
}

// ---------------- launch ----------------
extern "C" void kernel_launch(void* const* d_in, const int* in_sizes, int n_in,
                              void* d_out, int out_size, void* d_ws, size_t ws_size,
                              hipStream_t stream) {
  (void)in_sizes; (void)n_in; (void)out_size; (void)ws_size;
  const float* zi = (const float*)d_in[0];
  const float* zj = (const float*)d_in[1];

  _Float16* Zhi    = (_Float16*)d_ws;                       // 5120*512 halfs
  _Float16* Zlo    = Zhi + (size_t)NROWS * KDIM;            // 5120*512 halfs
  float*    sumexp = (float*)(Zlo + (size_t)NROWS * KDIM);  // 5120 floats
  float*    pos    = sumexp + NROWS;                        // 5120 floats
  float*    out    = (float*)d_out;

  hipLaunchKernelGGL(simclr_init_kernel, dim3(NROWS / 256), dim3(256), 0, stream,
                     sumexp, pos);
  hipLaunchKernelGGL(simclr_pool_norm_kernel, dim3(NROWS), dim3(256), 0, stream,
                     zi, zj, Zhi, Zlo);
  hipLaunchKernelGGL(simclr_gemm_kernel, dim3(NTILE), dim3(256), 0, stream,
                     Zhi, Zlo, sumexp, pos);
  hipLaunchKernelGGL(simclr_finalize_kernel, dim3(1), dim3(256), 0, stream,
                     sumexp, pos, out);
}
